// GCN_wCB_43834436223507
// MI455X (gfx1250) — matrix operand order repaired and checked by reading, not compiled, not run on hardware
//
#include <hip/hip_runtime.h>
#include <hip/hip_bf16.h>

// GCN layer: Out[b] = (A @ Z[b]) @ W  ==  A @ (Z[b] @ W)
// B=16, N=4096, D_IN=D_OUT=64.
// Kernel 1: Yt[b] = (Z[b] @ W)^T in bf16  (WMMA, Wt-fragments hoisted)
// Kernel 2: Out[b] = A @ Y[b]             (WMMA bf16, f32 accumulate)
//   Block = 16 waves covering 4 batches x 4 row-tiles: waves sharing a batch
//   reuse B-fragments via WGP$, waves sharing a row-tile reuse the A stripe.
//   L2 traffic ~ 256 MB (A) + 512 MB (B) vs 2 GB for the naive batch-per-wave map.

#define BATCH 16
#define NN    4096
#define DD    64

typedef __attribute__((ext_vector_type(16))) __bf16 v16bf;
typedef __attribute__((ext_vector_type(8)))  __bf16 v8bf;
typedef __attribute__((ext_vector_type(8)))  float  v8f;
typedef __attribute__((ext_vector_type(4)))  float  v4f;

static __device__ __forceinline__ v16bf cvt16(v4f a, v4f b, v4f c, v4f d) {
  v16bf r;
  r[0]  = (__bf16)a.x; r[1]  = (__bf16)a.y; r[2]  = (__bf16)a.z; r[3]  = (__bf16)a.w;
  r[4]  = (__bf16)b.x; r[5]  = (__bf16)b.y; r[6]  = (__bf16)b.z; r[7]  = (__bf16)b.w;
  r[8]  = (__bf16)c.x; r[9]  = (__bf16)c.y; r[10] = (__bf16)c.z; r[11] = (__bf16)c.w;
  r[12] = (__bf16)d.x; r[13] = (__bf16)d.y; r[14] = (__bf16)d.z; r[15] = (__bf16)d.w;
  return r;
}

// ---------------- Kernel 1: Yt[b][j][i] = sum_d Z[b][i][d] * W[d][j]  (bf16) -------
// One wave per (b, i-tile of 16). D = Wt(16x32) x Zt(32x16) over K=64 (2 k-steps).
__global__ __launch_bounds__(128) void zw_to_yt(const float* __restrict__ Z,
                                                const float* __restrict__ W,
                                                __bf16* __restrict__ Yt) {
  const int lane  = threadIdx.x & 31;
  const int lo    = lane & 15;
  const int hi    = lane >> 4;
  const int wave  = blockIdx.x * (blockDim.x >> 5) + (threadIdx.x >> 5); // 0..4095
  const int b     = wave >> 8;          // batch
  const int i0    = (wave & 255) << 4;  // i-tile origin
  const int iL    = i0 + lo;

  // A-fragments = W^T tiles (row j, K = d). 4 j-tiles x 2 k-steps, hoisted.
  v16bf wf[8];
#pragma unroll
  for (int jt = 0; jt < 4; ++jt) {
#pragma unroll
    for (int ks = 0; ks < 2; ++ks) {
      const int j    = jt * 16 + lo;
      const int base = ks * 32 + (hi << 3);
      v16bf f;
#pragma unroll
      for (int e = 0; e < 8; ++e)  f[e] = (__bf16)W[(base + e) * DD + j];       // K=base..base+7
#pragma unroll
      for (int e = 8; e < 16; ++e) f[e] = (__bf16)W[(base + e + 8) * DD + j];   // K=base+16..+23
      wf[jt * 2 + ks] = f;
    }
  }

  v8f acc[4] = {};
  const float* zrow = Z + ((size_t)(b * NN) + iL) * DD;   // B-frag: col i=iL, K=d contiguous
#pragma unroll
  for (int ks = 0; ks < 2; ++ks) {
    const int db = ks * 32 + (hi << 3);
    const v4f* zp = reinterpret_cast<const v4f*>(zrow + db);
    const v4f* zq = reinterpret_cast<const v4f*>(zrow + db + 16);
    v16bf bfrag = cvt16(zp[0], zp[1], zq[0], zq[1]);
#pragma unroll
    for (int jt = 0; jt < 4; ++jt) {
      acc[jt] = __builtin_amdgcn_wmma_f32_16x16x32_bf16(
          false, wf[jt * 2 + ks], false, bfrag, (short)0, acc[jt], false, false);
    }
  }

  // D layout: col N=lo (our i), row M = v + 8*hi (our j within tile)
#pragma unroll
  for (int jt = 0; jt < 4; ++jt) {
#pragma unroll
    for (int v = 0; v < 8; ++v) {
      const int j = jt * 16 + v + (hi << 3);
      Yt[((size_t)(b * DD) + j) * NN + iL] = (__bf16)acc[jt][v];
    }
  }
}

// ---------------- Kernel 2: Out[b] = A @ Y[b] -------------------------------------
// Block = 16 waves = 4 batches x 4 row-tiles. Wave computes a 16x64 output tile.
__global__ __launch_bounds__(512) void a_times_y(const float* __restrict__ A,
                                                 const __bf16* __restrict__ Yt,
                                                 float* __restrict__ out) {
  const int lane = threadIdx.x & 31;
  const int lo   = lane & 15;
  const int hi   = lane >> 4;
  const int w    = threadIdx.x >> 5;                 // 0..15
  const int bg   = blockIdx.x & 3;                   // batch group (4 batches each)
  const int rg   = blockIdx.x >> 2;                  // row group   (4 row-tiles each)
  const int b    = bg * 4 + (w & 3);                 // this wave's batch
  const int m0   = (rg * 4 + (w >> 2)) << 4;         // this wave's row-tile origin

  const float* arow = A + (size_t)(m0 + lo) * NN;

  v8f acc[4] = {};
  for (int k = 0; k < NN; k += 32) {
    // A-frag: row m0+lo; K = koff..koff+7 (e0-7), koff+16..+23 (e8-15)
    const int koff = k + (hi << 3);
    const v4f* ap0 = reinterpret_cast<const v4f*>(arow + koff);
    const v4f* ap1 = reinterpret_cast<const v4f*>(arow + koff + 16);
    v16bf afrag = cvt16(ap0[0], ap0[1], ap1[0], ap1[1]);

    const int kb = k + (hi << 3);          // B-frag: col n, K=kb..kb+7 (e0-7), kb+16..+23 (e8-15) in Yt
#pragma unroll
    for (int nt = 0; nt < 4; ++nt) {
      const v8bf* bp = reinterpret_cast<const v8bf*>(
          Yt + ((size_t)(b * DD + nt * 16 + lo)) * NN + kb);
      v8bf b0 = bp[0], b1 = bp[2];
      v16bf bfrag = __builtin_shufflevector(b0, b1, 0, 1, 2, 3, 4, 5, 6, 7,
                                            8, 9, 10, 11, 12, 13, 14, 15);
      acc[nt] = __builtin_amdgcn_wmma_f32_16x16x32_bf16(
          false, afrag, false, bfrag, (short)0, acc[nt], false, false);
    }
  }

  // D layout: col = nt*16 + lo, row = m0 + v + 8*hi
#pragma unroll
  for (int nt = 0; nt < 4; ++nt) {
#pragma unroll
    for (int v = 0; v < 8; ++v) {
      const int row = m0 + v + (hi << 3);
      out[((size_t)(b * NN) + row) * DD + nt * 16 + lo] = acc[nt][v];
    }
  }
}

extern "C" void kernel_launch(void* const* d_in, const int* in_sizes, int n_in,
                              void* d_out, int out_size, void* d_ws, size_t ws_size,
                              hipStream_t stream) {
  const float* Z = (const float*)d_in[0];   // [16][4096][64] f32
  const float* A = (const float*)d_in[1];   // [4096][4096]   f32
  const float* W = (const float*)d_in[2];   // [64][64]       f32
  float*     out = (float*)d_out;           // [16][4096][64] f32
  __bf16*     Yt = (__bf16*)d_ws;           // [16][64][4096] bf16 = 8 MB scratch

  // Kernel 1: 4096 waves (one per (b, i-tile)), 4 waves/block
  zw_to_yt<<<1024, 128, 0, stream>>>(Z, W, Yt);
  // Kernel 2: 256 blocks (4 batch-groups x 64 row-groups), 16 waves/block
  a_times_y<<<256, 512, 0, stream>>>(A, Yt, out);
}
